// StreamingVocos_1159641170336
// MI455X (gfx1250) — matrix operand order repaired and checked by reading, not compiled, not run on hardware
//
#include <hip/hip_runtime.h>
#include <math.h>

// ---------------------------------------------------------------------------
// Types for WMMA fragments (gfx1250, wave32)
// ---------------------------------------------------------------------------
typedef __attribute__((ext_vector_type(16))) __bf16 v16bf;
typedef __attribute__((ext_vector_type(8)))  float  v8f;
typedef __attribute__((ext_vector_type(4)))  int    v4i_;

union FragBF { v16bf v; uint4 q[2]; };

__device__ __forceinline__ unsigned short f2bf(float f) {
    unsigned u = __float_as_uint(f);
    unsigned r = u + 0x7FFFu + ((u >> 16) & 1u);   // RNE
    return (unsigned short)(r >> 16);
}

#define PI_F 3.14159265358979323846f

// ---------------------------------------------------------------------------
// CDNA5 async global->LDS copy (ASYNCcnt-tracked), 16 bytes per op.
// Builtin signature (probed from diagnostics): (int4 AS1*, int4 AS3*, Ii, Ii)
// ---------------------------------------------------------------------------
__device__ __forceinline__ void async_cp16(const unsigned short* g, unsigned short* l) {
#if defined(__has_builtin) && __has_builtin(__builtin_amdgcn_global_load_async_to_lds_b128)
    __builtin_amdgcn_global_load_async_to_lds_b128(
        (__attribute__((address_space(1))) v4i_*)g,
        (__attribute__((address_space(3))) v4i_*)l, 0, 0);
#else
    unsigned loff = (unsigned)(uintptr_t)l;     // low 32 bits = LDS byte offset
    asm volatile("global_load_async_to_lds_b128 %0, %1, off"
                 :: "v"(loff), "v"(g) : "memory");
#endif
}

__device__ __forceinline__ void wait_async0() {
#if defined(__has_builtin) && __has_builtin(__builtin_amdgcn_s_wait_asynccnt)
    __builtin_amdgcn_s_wait_asynccnt(0);
#else
    asm volatile("s_wait_asynccnt 0x0" ::: "memory");
#endif
}

// ---------------------------------------------------------------------------
// Workgroup-cooperative WMMA GEMM:  C[M x N] = A[M x K] * Bt[K x N]
//   A row-major bf16, Bt row-major bf16 (pre-transposed weights).
//   WG tile: 128(M) x 128(N); 8 waves as 4(M) x 2(N), each 32x64 (2x4 frags).
//   K stepped by 32; A(128x32) and B(32x128) tiles double-buffered in LDS,
//   filled with global_load_async_to_lds_b128 and read via ds_load_b128 (A) and per-element 16-bit loads (B).
//   Requires: M % 128 == 0, N % 128 == 0, K % 32 == 0  (all true here).
// MODE 0: Cf = acc + bias (f32)
// MODE 1: Cbf = bf16( gelu_exact(acc + bias) )
// MODE 2: Cf += gamma[n] * (acc + bias)      (in-place residual)
// ---------------------------------------------------------------------------
template <int MODE>
__global__ __launch_bounds__(256) void k_gemm(
    const unsigned short* __restrict__ A, const unsigned short* __restrict__ Bt,
    float* __restrict__ Cf, unsigned short* __restrict__ Cbf,
    const float* __restrict__ bias, const float* __restrict__ gamma,
    int N, int K, int Nreal, int ntn)
{
    __shared__ __align__(16) unsigned short sA[2][128 * 32];   // 8 KB each
    __shared__ __align__(16) unsigned short sB[2][32 * 128];   // 8 KB each

    const int tid  = threadIdx.x;
    const int lane = tid & 31;
    const int wid  = tid >> 5;
    const int m0   = (blockIdx.x / ntn) * 128;
    const int n0   = (blockIdx.x % ntn) * 128;
    const int mw   = (wid >> 1) * 32;       // wave row offset in WG tile
    const int nw   = (wid & 1) * 64;        // wave col offset in WG tile

    const int rowA  = lane & 15;            // A fragment: lane -> row m
    const int khalf = (lane >> 4) << 3;     // lanes 16-31: K offset +8

    auto issue_tile = [&](int k0, int buf) {
        {   // A tile: 128 rows x 32 cols; chunk ci -> row=ci/4, col=(ci%4)*8
            int r0 = tid >> 2,         c0 = (tid & 3) << 3;
            int r1 = (tid + 256) >> 2, c1 = ((tid + 256) & 3) << 3;
            async_cp16(A + (size_t)(m0 + r0) * K + k0 + c0, &sA[buf][r0 * 32 + c0]);
            async_cp16(A + (size_t)(m0 + r1) * K + k0 + c1, &sA[buf][r1 * 32 + c1]);
        }
        {   // B tile: 32 rows x 128 cols; chunk ci -> row=ci/16, col=(ci%16)*8
            int r0 = tid >> 4,         c0 = (tid & 15) << 3;
            int r1 = (tid + 256) >> 4, c1 = ((tid + 256) & 15) << 3;
            async_cp16(Bt + (size_t)(k0 + r0) * N + n0 + c0, &sB[buf][r0 * 128 + c0]);
            async_cp16(Bt + (size_t)(k0 + r1) * N + n0 + c1, &sB[buf][r1 * 128 + c1]);
        }
    };

    v8f acc[2][4];
    #pragma unroll
    for (int i = 0; i < 2; ++i)
        #pragma unroll
        for (int j = 0; j < 4; ++j)
            #pragma unroll
            for (int r = 0; r < 8; ++r) acc[i][j][r] = 0.0f;

    const int nk = K >> 5;
    issue_tile(0, 0);

    for (int it = 0; it < nk; ++it) {
        const int buf = it & 1;
        wait_async0();          // own tile-`it` async ops complete
        __syncthreads();        // all waves' tile data valid; prev reads done
        if (it + 1 < nk) issue_tile((it + 1) << 5, buf ^ 1);

        FragBF a[2], b[4];
        #pragma unroll
        for (int i = 0; i < 2; ++i) {
            const unsigned short* p = &sA[buf][(mw + i * 16 + rowA) * 32 + khalf];
            a[i].q[0] = *(const uint4*)(p);         // K = khalf .. +7
            a[i].q[1] = *(const uint4*)(p + 16);    // K = khalf+16 .. +23
        }
        #pragma unroll
        for (int j = 0; j < 4; ++j) {
            const unsigned short* p = &sB[buf][khalf * 128 + nw + j * 16 + rowA];
            #pragma unroll
            for (int s = 0; s < 16; ++s)
                ((unsigned short*)&b[j])[s] = p[(s + (s >> 3) * 8) * 128];
        }
        #pragma unroll
        for (int i = 0; i < 2; ++i)
            #pragma unroll
            for (int j = 0; j < 4; ++j)
                acc[i][j] = __builtin_amdgcn_wmma_f32_16x16x32_bf16(
                    false, a[i].v, false, b[j].v, (short)0, acc[i][j], false, false);
    }

    // C fragment layout: VGPR r -> M = r (lanes 0-15) / r+8 (lanes 16-31); N = lane%16
    const int madd = (lane >> 4) * 8;
    const int ncol = lane & 15;
    #pragma unroll
    for (int i = 0; i < 2; ++i) {
        #pragma unroll
        for (int j = 0; j < 4; ++j) {
            int nn = n0 + nw + j * 16 + ncol;
            float bz = bias ? ((nn < Nreal) ? bias[nn] : 0.0f) : 0.0f;
            float gm = (MODE == 2) ? gamma[nn] : 0.0f;
            #pragma unroll
            for (int r = 0; r < 8; ++r) {
                int mm = m0 + mw + i * 16 + madd + r;
                float v = acc[i][j][r] + bz;
                size_t idx = (size_t)mm * N + nn;
                if (MODE == 0) {
                    Cf[idx] = v;
                } else if (MODE == 1) {
                    float ge = 0.5f * v * (1.0f + erff(v * 0.70710678118f));
                    Cbf[idx] = f2bf(ge);
                } else {
                    Cf[idx] = Cf[idx] + gm * v;
                }
            }
        }
    }
}

// ---------------------------------------------------------------------------
// Weight transpose + f32->bf16:  Bt[k, n] = W[n, k]  (zero-padded to Kp x Np)
// ---------------------------------------------------------------------------
__global__ __launch_bounds__(256) void k_w2bt(
    const float* __restrict__ W, unsigned short* __restrict__ Bt,
    int N, int K, int Np, int Kp)
{
    size_t id = (size_t)blockIdx.x * 256 + threadIdx.x;
    if (id >= (size_t)Np * Kp) return;
    int n = (int)(id % Np);
    int k = (int)(id / Np);
    float v = (n < N && k < K) ? W[(size_t)n * K + k] : 0.0f;
    Bt[(size_t)k * Np + n] = f2bf(v);
}

// ---------------------------------------------------------------------------
// iRFFT * hann-window basis, bf16, K x N = 1056 x 1024 (rows >=1026 zero)
// ---------------------------------------------------------------------------
__global__ __launch_bounds__(256) void k_basis(unsigned short* __restrict__ Bt)
{
    size_t id = (size_t)blockIdx.x * 256 + threadIdx.x;
    if (id >= 1056UL * 1024UL) return;
    int n = (int)(id & 1023);
    int k = (int)(id >> 10);
    float w = 0.5f * (1.0f - cosf(2.0f * PI_F * (float)n * (1.0f / 1024.0f)));
    float v = 0.0f;
    if (k < 513) {
        int f = k;
        float a = (f == 0 || f == 512) ? 1.0f : 2.0f;
        int p = (f * n) & 1023;
        v = a * cosf(2.0f * PI_F * (float)p * (1.0f / 1024.0f)) * w * (1.0f / 1024.0f);
    } else if (k < 1026) {
        int f = k - 513;
        if (f != 0 && f != 512) {
            int p = (f * n) & 1023;
            v = -2.0f * sinf(2.0f * PI_F * (float)p * (1.0f / 1024.0f)) * w * (1.0f / 1024.0f);
        }
    }
    Bt[(size_t)k * 1024 + n] = f2bf(v);
}

// ---------------------------------------------------------------------------
// im2col for embed conv: Xc[(b,t), c*7+k] = mel[b, c, t-6+k]  (causal, 704 pad)
// ---------------------------------------------------------------------------
__global__ __launch_bounds__(256) void k_im2col(
    const float* __restrict__ mel, unsigned short* __restrict__ xc)
{
    size_t id = (size_t)blockIdx.x * 256 + threadIdx.x;
    if (id >= 8192UL * 704UL) return;
    int col = (int)(id % 704);
    size_t m = id / 704;
    int bb = (int)(m >> 10);
    int t  = (int)(m & 1023);
    float v = 0.0f;
    if (col < 700) {
        int c = col / 7, k = col % 7;
        int tt = t - 6 + k;
        if (tt >= 0) v = mel[((size_t)bb * 100 + c) * 1024 + tt];
    }
    xc[m * 704 + col] = f2bf(v);
}

// ---------------------------------------------------------------------------
// LayerNorm over DIM=512: one wave per row; optional f32 and bf16 outputs.
// ---------------------------------------------------------------------------
__global__ __launch_bounds__(256) void k_ln(
    const float* __restrict__ X, const float* __restrict__ g, const float* __restrict__ b,
    float* __restrict__ outf, unsigned short* __restrict__ outb, int rows)
{
    int wave = blockIdx.x * (blockDim.x >> 5) + (threadIdx.x >> 5);
    if (wave >= rows) return;
    int lane = threadIdx.x & 31;
    const float* xr = X + (size_t)wave * 512;
    float v[16], s = 0.0f, s2 = 0.0f;
    #pragma unroll
    for (int i = 0; i < 16; ++i) {
        float t = xr[lane + 32 * i];
        v[i] = t; s += t; s2 += t * t;
    }
    #pragma unroll
    for (int o = 16; o > 0; o >>= 1) {
        s  += __shfl_xor(s,  o, 32);
        s2 += __shfl_xor(s2, o, 32);
    }
    float mu  = s * (1.0f / 512.0f);
    float var = s2 * (1.0f / 512.0f) - mu * mu;
    float inv = rsqrtf(var + 1e-6f);
    #pragma unroll
    for (int i = 0; i < 16; ++i) {
        int d = lane + 32 * i;
        float y = (v[i] - mu) * inv * g[d] + b[d];
        if (outf) outf[(size_t)wave * 512 + d] = y;
        if (outb) outb[(size_t)wave * 512 + d] = f2bf(y);
    }
}

// ---------------------------------------------------------------------------
// Fused depthwise causal conv (K=7) + LayerNorm -> bf16 A operand
// ---------------------------------------------------------------------------
__global__ __launch_bounds__(256) void k_dwln(
    const float* __restrict__ x, const float* __restrict__ dw, const float* __restrict__ dwb,
    const float* __restrict__ g, const float* __restrict__ b,
    unsigned short* __restrict__ outb)
{
    int wave = blockIdx.x * (blockDim.x >> 5) + (threadIdx.x >> 5);
    if (wave >= 8192) return;
    int lane = threadIdx.x & 31;
    int bb = wave >> 10;
    int t  = wave & 1023;
    float v[16], s = 0.0f, s2 = 0.0f;
    #pragma unroll
    for (int i = 0; i < 16; ++i) {
        int d = lane + 32 * i;
        float acc = dwb[d];
        #pragma unroll
        for (int k = 0; k < 7; ++k) {
            int tt = t - 6 + k;
            if (tt >= 0) acc += x[((size_t)(bb << 10) + tt) * 512 + d] * dw[d * 7 + k];
        }
        v[i] = acc; s += acc; s2 += acc * acc;
    }
    #pragma unroll
    for (int o = 16; o > 0; o >>= 1) {
        s  += __shfl_xor(s,  o, 32);
        s2 += __shfl_xor(s2, o, 32);
    }
    float mu  = s * (1.0f / 512.0f);
    float var = s2 * (1.0f / 512.0f) - mu * mu;
    float inv = rsqrtf(var + 1e-6f);
    #pragma unroll
    for (int i = 0; i < 16; ++i) {
        int d = lane + 32 * i;
        float y = (v[i] - mu) * inv * g[d] + b[d];
        outb[(size_t)wave * 512 + d] = f2bf(y);
    }
}

// ---------------------------------------------------------------------------
// mag/phase -> packed [Re | Im | 0pad] bf16 matrix G (8192 x 1056)
// Hout stride 1152 (head GEMM padded N).
// ---------------------------------------------------------------------------
__global__ __launch_bounds__(256) void k_magphase(
    const float* __restrict__ hout, unsigned short* __restrict__ G)
{
    size_t id = (size_t)blockIdx.x * 256 + threadIdx.x;
    if (id >= 8192UL * 544UL) return;
    int f = (int)(id % 544);
    size_t m = id / 544;
    if (f < 513) {
        float mg = fminf(expf(hout[m * 1152 + f]), 100.0f);
        float ph = hout[m * 1152 + 513 + f];
        G[m * 1056 + f]       = f2bf(mg * cosf(ph));
        G[m * 1056 + 513 + f] = f2bf(mg * sinf(ph));
    } else {
        int c = 513 + f;                 // 1026 .. 1055
        if (c < 1056) G[m * 1056 + c] = 0;
    }
}

// ---------------------------------------------------------------------------
// Overlap-add + warm-up envelope normalization -> audio (B, T*HOP) f32
// ---------------------------------------------------------------------------
__global__ __launch_bounds__(256) void k_ola(
    const float* __restrict__ frames, float* __restrict__ audio)
{
    size_t id = (size_t)blockIdx.x * 256 + threadIdx.x;
    if (id >= 8UL * 1024UL * 256UL) return;
    int h  = (int)(id & 255);
    int t  = (int)((id >> 8) & 1023);
    int bb = (int)(id >> 18);
    float out = 0.0f, env = 0.0f;
    #pragma unroll
    for (int k = 0; k < 4; ++k) {
        int tf = t - k;
        if (tf >= 0) out += frames[((size_t)(bb << 10) + tf) * 1024 + (k << 8) + h];
        if (k <= t) {
            int n = (k << 8) + h;
            float w = 0.5f * (1.0f - cosf(2.0f * PI_F * (float)n * (1.0f / 1024.0f)));
            env += w * w;
        }
    }
    audio[id] = out / fmaxf(env, 1e-8f);
}

// ---------------------------------------------------------------------------
// Host orchestration
// ---------------------------------------------------------------------------
extern "C" void kernel_launch(void* const* d_in, const int* in_sizes, int n_in,
                              void* d_out, int out_size, void* d_ws, size_t ws_size,
                              hipStream_t stream)
{
    (void)in_sizes; (void)n_in; (void)out_size; (void)ws_size;

    const float* mel     = (const float*)d_in[0];
    const float* embed_w = (const float*)d_in[1];   // (512, 100, 7) == (512, 700)
    const float* embed_b = (const float*)d_in[2];
    const float* in_g    = (const float*)d_in[3];
    const float* in_b    = (const float*)d_in[4];
    // blocks: base 5 + 9*i : dw_w, dw_b, ln_g, ln_b, w1, b1, w2, b2, gamma
    const float* final_g = (const float*)d_in[77];
    const float* final_b = (const float*)d_in[78];
    const float* head_w  = (const float*)d_in[79];  // (1026, 512)
    const float* head_b  = (const float*)d_in[80];

    char* ws = (char*)d_ws;
    size_t off = 0;
    auto alloc = [&](size_t bytes) -> char* {
        char* p = ws + off;
        off = (off + bytes + 255) & ~(size_t)255;
        return p;
    };

    unsigned short* bt_embed = (unsigned short*)alloc(704UL * 512 * 2);
    unsigned short* bt_w1[8];
    unsigned short* bt_w2[8];
    for (int i = 0; i < 8; ++i) bt_w1[i] = (unsigned short*)alloc(512UL * 1536 * 2);
    for (int i = 0; i < 8; ++i) bt_w2[i] = (unsigned short*)alloc(1536UL * 512 * 2);
    unsigned short* bt_head  = (unsigned short*)alloc(512UL * 1152 * 2);
    unsigned short* bt_basis = (unsigned short*)alloc(1056UL * 1024 * 2);
    float*          x        = (float*)alloc(8192UL * 512 * 4);
    unsigned short* abf      = (unsigned short*)alloc(8192UL * 512 * 2);
    unsigned short* hbf      = (unsigned short*)alloc(8192UL * 1536 * 2);
    float*          hout     = (float*)alloc(8192UL * 1152 * 4);
    unsigned short* G        = (unsigned short*)alloc(8192UL * 1056 * 2);
    float*          frames   = (float*)alloc(8192UL * 1024 * 4);
    unsigned short* xc       = (unsigned short*)frames;   // alias: dead before frames

    auto blks = [](size_t n) { return dim3((unsigned)((n + 255) / 256)); };
    const dim3 tpb(256);

    // --- weight preprocessing (bf16, transposed to K x N) ---
    hipLaunchKernelGGL(k_w2bt, blks(704UL * 512), tpb, 0, stream,
                       embed_w, bt_embed, 512, 700, 512, 704);
    for (int i = 0; i < 8; ++i) {
        const float* w1 = (const float*)d_in[5 + 9 * i + 4];   // (1536, 512)
        const float* w2 = (const float*)d_in[5 + 9 * i + 6];   // (512, 1536)
        hipLaunchKernelGGL(k_w2bt, blks(512UL * 1536), tpb, 0, stream,
                           w1, bt_w1[i], 1536, 512, 1536, 512);
        hipLaunchKernelGGL(k_w2bt, blks(1536UL * 512), tpb, 0, stream,
                           w2, bt_w2[i], 512, 1536, 512, 1536);
    }
    hipLaunchKernelGGL(k_w2bt, blks(512UL * 1152), tpb, 0, stream,
                       head_w, bt_head, 1026, 512, 1152, 512);
    hipLaunchKernelGGL(k_basis, blks(1056UL * 1024), tpb, 0, stream, bt_basis);

    // --- embed conv as GEMM: x = Xc(8192x704) * BtE(704x512) + b ---
    hipLaunchKernelGGL(k_im2col, blks(8192UL * 704), tpb, 0, stream, mel, xc);
    {
        int ntn = 512 / 128;
        hipLaunchKernelGGL(HIP_KERNEL_NAME(k_gemm<0>), dim3(64 * ntn), tpb, 0, stream,
                           xc, bt_embed, x, (unsigned short*)nullptr,
                           embed_b, (const float*)nullptr, 512, 704, 512, ntn);
    }
    // in_norm (in-place f32)
    hipLaunchKernelGGL(k_ln, dim3(8192 / 8), tpb, 0, stream,
                       x, in_g, in_b, x, (unsigned short*)nullptr, 8192);

    // --- 8 ConvNeXt blocks ---
    for (int i = 0; i < 8; ++i) {
        const float* dw_w  = (const float*)d_in[5 + 9 * i + 0];
        const float* dw_b  = (const float*)d_in[5 + 9 * i + 1];
        const float* ln_g  = (const float*)d_in[5 + 9 * i + 2];
        const float* ln_b  = (const float*)d_in[5 + 9 * i + 3];
        const float* b1    = (const float*)d_in[5 + 9 * i + 5];
        const float* b2    = (const float*)d_in[5 + 9 * i + 7];
        const float* gamma = (const float*)d_in[5 + 9 * i + 8];

        hipLaunchKernelGGL(k_dwln, dim3(8192 / 8), tpb, 0, stream,
                           x, dw_w, dw_b, ln_g, ln_b, abf);
        {   // FFN1: h = gelu(a * w1^T + b1) -> bf16
            int ntn = 1536 / 128;
            hipLaunchKernelGGL(HIP_KERNEL_NAME(k_gemm<1>), dim3(64 * ntn), tpb, 0, stream,
                               abf, bt_w1[i], (float*)nullptr, hbf,
                               b1, (const float*)nullptr, 1536, 512, 1536, ntn);
        }
        {   // FFN2: x += gamma * (h * w2^T + b2)
            int ntn = 512 / 128;
            hipLaunchKernelGGL(HIP_KERNEL_NAME(k_gemm<2>), dim3(64 * ntn), tpb, 0, stream,
                               hbf, bt_w2[i], x, (unsigned short*)nullptr,
                               b2, gamma, 512, 1536, 512, ntn);
        }
    }

    // --- final LN -> bf16 A ---
    hipLaunchKernelGGL(k_ln, dim3(8192 / 8), tpb, 0, stream,
                       x, final_g, final_b, (float*)nullptr, abf, 8192);

    // --- head GEMM (N padded 1026 -> 1152 = 9 x 128) ---
    {
        int ntn = 1152 / 128;
        hipLaunchKernelGGL(HIP_KERNEL_NAME(k_gemm<0>), dim3(64 * ntn), tpb, 0, stream,
                           abf, bt_head, hout, (unsigned short*)nullptr,
                           head_b, (const float*)nullptr, 1152, 512, 1026, ntn);
    }

    // --- spectrum -> [Re|Im] bf16 ---
    hipLaunchKernelGGL(k_magphase, blks(8192UL * 544), tpb, 0, stream, hout, G);

    // --- windowed iRFFT as GEMM: frames = G(8192x1056) * basis(1056x1024) ---
    {
        int ntn = 1024 / 128;
        hipLaunchKernelGGL(HIP_KERNEL_NAME(k_gemm<0>), dim3(64 * ntn), tpb, 0, stream,
                           G, bt_basis, frames, (unsigned short*)nullptr,
                           (const float*)nullptr, (const float*)nullptr,
                           1024, 1056, 1024, ntn);
    }

    // --- overlap-add + envelope ---
    hipLaunchKernelGGL(k_ola, blks(8UL * 1024 * 256), tpb, 0, stream,
                       frames, (float*)d_out);
}
